// MultiHeadAttention_47949014892945
// MI455X (gfx1250) — matrix operand order repaired and checked by reading, not compiled, not run on hardware
//
#include <hip/hip_runtime.h>
#include <hip/hip_bf16.h>

typedef __attribute__((ext_vector_type(16))) __bf16 v16bf;
typedef __attribute__((ext_vector_type(8)))  float  v8f;

#define NEGV (-1.0e12f)

union Frag {
    v16bf v;
    unsigned int u[8];
    uint4 q[2];
};

static __device__ __forceinline__ v8f wmma_bf16(v16bf a, v16bf b, v8f c) {
    // D = A(16x32 bf16) * B(32x16 bf16) + C(16x16 f32)
    return __builtin_amdgcn_wmma_f32_16x16x32_bf16(
        /*neg_a=*/false, a, /*neg_b=*/false, b,
        /*c_mod=*/(short)0, c, /*reuse_a=*/false, /*reuse_b=*/false);
}

static __device__ __forceinline__ unsigned int pk_bf16(float x, float y) {
    const __bf16 a = (__bf16)x, b = (__bf16)y;
    const unsigned short ua = __builtin_bit_cast(unsigned short, a);
    const unsigned short ub = __builtin_bit_cast(unsigned short, b);
    return (unsigned int)ua | ((unsigned int)ub << 16);
}

// A-fragment packed-u32 index for the ISA 16-bit 16x32 A layout:
// u[j] holds K pair {base+2j} (j<4) / {base+8+2j} (j>=4), base = (lane>>4)*8
static __device__ __forceinline__ int afrag_uidx(int lane, int j) {
    return ((lane >> 4) << 2) + j + ((j >> 2) << 2);
}

// ---------------------------------------------------------------------------
// Projection GEMM: Y(bf16)[M,N] = X(f32)[M,K] @ W(f32)[N,K]^T + bias
// block = 256 (8 waves); block tile 64x128; wave tile 32x32 (4 WMMA / k-step)
// grid = (M/64, N/128)
// ---------------------------------------------------------------------------
__global__ __launch_bounds__(256) void k_proj(const float* __restrict__ X,
                                              const float* __restrict__ W,
                                              const float* __restrict__ bias,
                                              __bf16* __restrict__ Y,
                                              int M, int N, int K) {
    __shared__ unsigned int axs[64 * 16];   // A tile 64x32 bf16 (4 KB)
    __shared__ unsigned int bxs[128 * 16];  // B tile 128x32 bf16 (8 KB)

    const int tid  = threadIdx.x;
    const int lane = tid & 31;
    const int wave = tid >> 5;
    const int wm   = wave >> 2;   // 0..1 -> 32-row group
    const int wn   = wave & 3;    // 0..3 -> 32-col group
    const int m0   = blockIdx.x * 64;
    const int n0   = blockIdx.y * 128;
    const int cl   = lane & 15;

    v8f acc00 = {}, acc01 = {}, acc10 = {}, acc11 = {};

    for (int k0 = 0; k0 < K; k0 += 32) {
        // ---- stage A tile (f32 -> bf16): 8 consecutive cols per thread ----
        {
            const int r  = tid >> 2;
            const int cb = (tid & 3) * 8;
            const float4* xp =
                reinterpret_cast<const float4*>(&X[(size_t)(m0 + r) * K + k0 + cb]);
            __builtin_prefetch(xp + 8, 0, 3);   // next k-step
            const float4 f0 = xp[0], f1 = xp[1];
            unsigned int* dst = &axs[r * 16 + (cb >> 1)];
            dst[0] = pk_bf16(f0.x, f0.y);
            dst[1] = pk_bf16(f0.z, f0.w);
            dst[2] = pk_bf16(f1.x, f1.y);
            dst[3] = pk_bf16(f1.z, f1.w);
        }
        // ---- stage B tile (f32 -> bf16): 16 consecutive cols per thread ----
        {
            const int n  = tid >> 1;
            const int cb = (tid & 1) * 16;
            const float4* wp =
                reinterpret_cast<const float4*>(&W[(size_t)(n0 + n) * K + k0 + cb]);
            __builtin_prefetch(wp + 8, 0, 3);
            unsigned int* dst = &bxs[n * 16 + (cb >> 1)];
#pragma unroll
            for (int t = 0; t < 4; ++t) {
                const float4 f = wp[t];
                dst[t * 2]     = pk_bf16(f.x, f.y);
                dst[t * 2 + 1] = pk_bf16(f.z, f.w);
            }
        }
        __syncthreads();

        Frag a0, a1, b0, b1;
#pragma unroll
        for (int j = 0; j < 8; ++j) {
            const int ai = afrag_uidx(lane, j);
            const int bi = afrag_uidx(lane, j);
            a0.u[j] = axs[(wm * 32 + cl) * 16 + ai];
            a1.u[j] = axs[(wm * 32 + 16 + cl) * 16 + ai];
            b0.u[j] = bxs[(wn * 32 + cl) * 16 + bi];
            b1.u[j] = bxs[(wn * 32 + 16 + cl) * 16 + bi];
        }
        acc00 = wmma_bf16(a0.v, b0.v, acc00);
        acc01 = wmma_bf16(a0.v, b1.v, acc01);
        acc10 = wmma_bf16(a1.v, b0.v, acc10);
        acc11 = wmma_bf16(a1.v, b1.v, acc11);
        __syncthreads();
    }

    // ---- epilogue: C/D layout -> row (lane>>4)*8 + r, col lane&15 ----
    const int rh = (lane >> 4) << 3;
    auto store_tile = [&](v8f acc, int rb, int col) {
        const float bia = bias[col];
#pragma unroll
        for (int r = 0; r < 8; ++r)
            Y[(size_t)(rb + r) * N + col] = (__bf16)(acc[r] + bia);
    };
    store_tile(acc00, m0 + wm * 32 + rh,      n0 + wn * 32 + cl);
    store_tile(acc01, m0 + wm * 32 + rh,      n0 + wn * 32 + 16 + cl);
    store_tile(acc10, m0 + wm * 32 + 16 + rh, n0 + wn * 32 + cl);
    store_tile(acc11, m0 + wm * 32 + 16 + rh, n0 + wn * 32 + 16 + cl);
}

// ---------------------------------------------------------------------------
// Fused attention: scores (WMMA, causal-skipped) -> masked softmax (LDS)
//                  -> attn output + P@V (WMMA)
// grid = (64 q-tiles, H=16, B=4), block = 256
// ---------------------------------------------------------------------------
__global__ __launch_bounds__(256) void k_attn(const __bf16* __restrict__ Qb,
                                              const __bf16* __restrict__ Kb,
                                              const __bf16* __restrict__ Vb,
                                              const int*    __restrict__ mask,
                                              float*  __restrict__ attn_out,
                                              __bf16* __restrict__ Ctx) {
    __shared__ unsigned int scu[16 * 512];   // scores -> probs, bf16 (32 KB)
    __shared__ float  red[16 * 16];
    __shared__ float  pacc[2][16 * 64];      // P@V partials per chunk-parity
    __shared__ unsigned char mk[1024];
    __shared__ unsigned char mq[16];
    __bf16* sc = reinterpret_cast<__bf16*>(scu);

    const int tid  = threadIdx.x;
    const int lane = tid & 31;
    const int wave = tid >> 5;
    const int qt = blockIdx.x;
    const int h  = blockIdx.y;
    const int b  = blockIdx.z;
    const int q0 = qt * 16;

    // stage masks
    {
        const int i4 = tid * 4;
#pragma unroll
        for (int j = 0; j < 4; ++j)
            mk[i4 + j] = (unsigned char)(mask[b * 1024 + i4 + j] != 0);
    }
    if (tid < 16) mq[tid] = (unsigned char)(mask[b * 1024 + q0 + tid] != 0);
    __syncthreads();

    const int mrow = lane & 15;
    const int bn   = lane & 15;
    const int rh   = (lane >> 4) << 3;

    // ---- Q A-fragments via packed u32 loads (reused for all key-tiles) ----
    Frag aq0, aq1;
    {
        const unsigned int* qu = reinterpret_cast<const unsigned int*>(
            &Qb[(size_t)(b * 1024 + q0 + mrow) * 1024 + h * 64]);
#pragma unroll
        for (int j = 0; j < 8; ++j) {
            const int idx = afrag_uidx(lane, j);
            aq0.u[j] = qu[idx];
            aq1.u[j] = qu[idx + 16];
        }
    }

    // ---- scores: wave w handles key-tiles w, w+8, ..., w+56 ----
    for (int j = 0; j < 8; ++j) {
        const int kt = wave + j * 8;
        const int c0 = kt * 16;
        if (c0 > q0 + 15) {           // fully in the future: no math needed
#pragma unroll
            for (int r = 0; r < 8; ++r)
                sc[(rh + r) * 1024 + c0 + bn] = (__bf16)NEGV;
            continue;
        }
        Frag b0, b1;
        const uint4* kq = reinterpret_cast<const uint4*>(
            &Kb[(size_t)(b * 1024 + c0 + bn) * 1024 + h * 64]);
        const int o = (lane >> 4);       // B-frag k-base in uint4 units
        b0.q[0] = kq[o];     b0.q[1] = kq[o + 2];   // d = 0..31
        b1.q[0] = kq[o + 4]; b1.q[1] = kq[o + 6];   // d = 32..63
        v8f s = {};
        s = wmma_bf16(aq0.v, b0.v, s);
        s = wmma_bf16(aq1.v, b1.v, s);
#pragma unroll
        for (int r = 0; r < 8; ++r) {
            const int row = rh + r;
            const int col = c0 + bn;
            float v = s[r] * 0.125f;   // 1/sqrt(64)
            if (mq[row] | mk[col] | (col > q0 + row)) v = NEGV;
            sc[row * 1024 + col] = (__bf16)v;
        }
    }
    __syncthreads();

    // ---- softmax: 16 threads per row; scans bounded by causal limit ----
    const int srow = tid >> 4;
    const int sl   = tid & 15;
    const int cmax = q0 + 16;          // multiple of 16
    float mx = -3.0e38f;
    for (int c = sl; c < cmax; c += 16)
        mx = fmaxf(mx, (float)sc[srow * 1024 + c]);
    red[srow * 16 + sl] = mx;
    __syncthreads();
    mx = red[srow * 16];
#pragma unroll
    for (int j = 1; j < 16; ++j) mx = fmaxf(mx, red[srow * 16 + j]);
    __syncthreads();

    float sum = 0.f;
    for (int c = sl; c < cmax; c += 16) {
        const float e = __expf((float)sc[srow * 1024 + c] - mx);
        sc[srow * 1024 + c] = (__bf16)e;
        sum += e;
    }
    red[srow * 16 + sl] = sum;
    __syncthreads();
    sum = 0.f;
#pragma unroll
    for (int j = 0; j < 16; ++j) sum += red[srow * 16 + j];
    const float inv = 1.0f / sum;

    // normalize full row: mandatory attn output + bf16 probs kept for P@V
    float* arow = &attn_out[((size_t)(h * 4 + b) * 1024 + q0 + srow) * 1024];
    for (int c = sl; c < 1024; c += 16) {
        float p = (float)sc[srow * 1024 + c] * inv;
        if (mq[srow] | mk[c] | (c > q0 + srow)) p = 0.f;
        sc[srow * 1024 + c] = (__bf16)p;
        arow[c] = p;
    }
    __syncthreads();

    // ---- P @ V over causal k-range only ----
    const int ntile   = wave & 3;
    const int par     = wave >> 2;
    const int nchunks = (q0 + 16 + 31) >> 5;
    v8f oa = {};
    for (int kc = par; kc < nchunks; kc += 2) {
        const int kk0 = kc * 32;
        Frag ap, bv;
#pragma unroll
        for (int j = 0; j < 8; ++j)
            ap.u[j] = scu[mrow * 512 + (kk0 >> 1) + ((lane >> 4) << 3) + j];
        const __bf16* vbase =
            &Vb[(size_t)(b * 1024 + kk0 + ((lane >> 4) << 4)) * 1024
                + h * 64 + ntile * 16 + bn];
#pragma unroll
        for (int i = 0; i < 16; ++i) bv.v[i] = vbase[(size_t)i * 1024];
        oa = wmma_bf16(ap.v, bv.v, oa);
    }
#pragma unroll
    for (int r = 0; r < 8; ++r)
        pacc[par][(rh + r) * 64 + ntile * 16 + bn] = oa[r];   // unique slot
    __syncthreads();

    // combine parities deterministically, write ctx (bf16)
    {
        const int r  = tid >> 4;
        const int d0 = (tid & 15) * 4;
        __bf16* crow = &Ctx[(size_t)(b * 1024 + q0 + r) * 1024 + h * 64 + d0];
#pragma unroll
        for (int j = 0; j < 4; ++j)
            crow[j] = (__bf16)(pacc[0][r * 64 + d0 + j] + pacc[1][r * 64 + d0 + j]);
    }
}

// ---------------------------------------------------------------------------
// Output projection: out(f32) = Ctx(bf16) @ Wo^T + bo + residual(query)
// same 64x128 block tile / 32x32 wave tile as k_proj
// ---------------------------------------------------------------------------
__global__ __launch_bounds__(256) void k_oproj(const __bf16* __restrict__ Xb,
                                               const float*  __restrict__ W,
                                               const float*  __restrict__ bias,
                                               const float*  __restrict__ resid,
                                               float* __restrict__ Y,
                                               int M, int N, int K) {
    __shared__ unsigned int axs[64 * 16];
    __shared__ unsigned int bxs[128 * 16];

    const int tid  = threadIdx.x;
    const int lane = tid & 31;
    const int wave = tid >> 5;
    const int wm   = wave >> 2;
    const int wn   = wave & 3;
    const int m0   = blockIdx.x * 64;
    const int n0   = blockIdx.y * 128;
    const int cl   = lane & 15;

    v8f acc00 = {}, acc01 = {}, acc10 = {}, acc11 = {};

    for (int k0 = 0; k0 < K; k0 += 32) {
        {   // stage A tile (already bf16): 16-byte copy per thread
            const int r  = tid >> 2;
            const int cb = (tid & 3) * 8;
            const uint4 v = *reinterpret_cast<const uint4*>(
                &Xb[(size_t)(m0 + r) * K + k0 + cb]);
            *reinterpret_cast<uint4*>(&axs[r * 16 + (cb >> 1)]) = v;
        }
        {   // stage B tile (f32 -> bf16)
            const int n  = tid >> 1;
            const int cb = (tid & 1) * 16;
            const float4* wp =
                reinterpret_cast<const float4*>(&W[(size_t)(n0 + n) * K + k0 + cb]);
            __builtin_prefetch(wp + 8, 0, 3);
            unsigned int* dst = &bxs[n * 16 + (cb >> 1)];
#pragma unroll
            for (int t = 0; t < 4; ++t) {
                const float4 f = wp[t];
                dst[t * 2]     = pk_bf16(f.x, f.y);
                dst[t * 2 + 1] = pk_bf16(f.z, f.w);
            }
        }
        __syncthreads();

        Frag a0, a1, b0, b1;
#pragma unroll
        for (int j = 0; j < 8; ++j) {
            const int ai = afrag_uidx(lane, j);
            const int bi = afrag_uidx(lane, j);
            a0.u[j] = axs[(wm * 32 + cl) * 16 + ai];
            a1.u[j] = axs[(wm * 32 + 16 + cl) * 16 + ai];
            b0.u[j] = bxs[(wn * 32 + cl) * 16 + bi];
            b1.u[j] = bxs[(wn * 32 + 16 + cl) * 16 + bi];
        }
        acc00 = wmma_bf16(a0.v, b0.v, acc00);
        acc01 = wmma_bf16(a0.v, b1.v, acc01);
        acc10 = wmma_bf16(a1.v, b0.v, acc10);
        acc11 = wmma_bf16(a1.v, b1.v, acc11);
        __syncthreads();
    }

    const int rh = (lane >> 4) << 3;
    auto store_tile = [&](v8f acc, int rb, int col) {
        const float bia = bias[col];
#pragma unroll
        for (int r = 0; r < 8; ++r) {
            const size_t idx = (size_t)(rb + r) * N + col;
            Y[idx] = acc[r] + bia + resid[idx];
        }
    };
    store_tile(acc00, m0 + wm * 32 + rh,      n0 + wn * 32 + cl);
    store_tile(acc01, m0 + wm * 32 + rh,      n0 + wn * 32 + 16 + cl);
    store_tile(acc10, m0 + wm * 32 + 16 + rh, n0 + wn * 32 + cl);
    store_tile(acc11, m0 + wm * 32 + 16 + rh, n0 + wn * 32 + 16 + cl);
}

// ---------------------------------------------------------------------------
// Row LayerNorm in place: one block (256 thr) per 1024-wide row
// ---------------------------------------------------------------------------
__global__ __launch_bounds__(256) void k_ln(float* __restrict__ Y,
                                            const float* __restrict__ g,
                                            const float* __restrict__ beta) {
    __shared__ float red[256];
    const int tid = threadIdx.x;
    float* row = &Y[(size_t)blockIdx.x * 1024];

    float v[4];
    float s = 0.f;
#pragma unroll
    for (int j = 0; j < 4; ++j) { v[j] = row[tid + j * 256]; s += v[j]; }
    red[tid] = s;
    __syncthreads();
    for (int off = 128; off > 0; off >>= 1) {
        if (tid < off) red[tid] += red[tid + off];
        __syncthreads();
    }
    const float mu = red[0] * (1.0f / 1024.0f);
    __syncthreads();

    float s2 = 0.f;
#pragma unroll
    for (int j = 0; j < 4; ++j) { const float d = v[j] - mu; s2 += d * d; }
    red[tid] = s2;
    __syncthreads();
    for (int off = 128; off > 0; off >>= 1) {
        if (tid < off) red[tid] += red[tid + off];
        __syncthreads();
    }
    const float rinv = rsqrtf(red[0] * (1.0f / 1024.0f) + 1e-12f);
#pragma unroll
    for (int j = 0; j < 4; ++j) {
        const int c = tid + j * 256;
        row[c] = (v[j] - mu) * rinv * g[c] + beta[c];
    }
}

// ---------------------------------------------------------------------------
extern "C" void kernel_launch(void* const* d_in, const int* in_sizes, int n_in,
                              void* d_out, int out_size, void* d_ws, size_t ws_size,
                              hipStream_t stream) {
    const float* query = (const float*)d_in[0];
    const float* key   = (const float*)d_in[1];
    const float* value = (const float*)d_in[2];
    const int*   mask  = (const int*)d_in[3];
    const float* Wq = (const float*)d_in[4];
    const float* bq = (const float*)d_in[5];
    const float* Wk = (const float*)d_in[6];
    const float* bk = (const float*)d_in[7];
    const float* Wv = (const float*)d_in[8];
    const float* bv = (const float*)d_in[9];
    const float* Wo = (const float*)d_in[10];
    const float* bo = (const float*)d_in[11];
    const float* ln_g = (const float*)d_in[12];
    const float* ln_b = (const float*)d_in[13];

    float* out  = (float*)d_out;                    // [4,1024,1024]
    float* attn = out + (size_t)4 * 1024 * 1024;    // [64,1024,1024]

    const size_t MK = (size_t)4096 * 1024;
    __bf16* Qb = (__bf16*)d_ws;
    __bf16* Kb = Qb + MK;
    __bf16* Vb = Kb + MK;
    __bf16* Cb = Vb + MK;                           // 32 MB total ws

    const dim3 gp(64, 8);    // 4096/64 m-tiles x 1024/128 n-tiles
    k_proj<<<gp, 256, 0, stream>>>(query, Wq, bq, Qb, 4096, 1024, 1024);
    k_proj<<<gp, 256, 0, stream>>>(key,   Wk, bk, Kb, 4096, 1024, 1024);
    k_proj<<<gp, 256, 0, stream>>>(value, Wv, bv, Vb, 4096, 1024, 1024);

    const dim3 ga(64, 16, 4);  // q-tile, head, batch
    k_attn<<<ga, 256, 0, stream>>>(Qb, Kb, Vb, mask, attn, Cb);

    k_oproj<<<gp, 256, 0, stream>>>(Cb, Wo, bo, query, out, 4096, 1024, 1024);
    k_ln<<<4096, 256, 0, stream>>>(out, ln_g, ln_b);
}
